// LlamaAttention_57844619542951
// MI455X (gfx1250) — matrix operand order repaired and checked by reading, not compiled, not run on hardware
//
#include <hip/hip_runtime.h>
#include <hip/hip_bf16.h>

// ---------------------------------------------------------------------------
// Llama attention for gfx1250 (MI455X): bf16 WMMA GEMMs + flash attention.
// Problem constants from the reference.
// ---------------------------------------------------------------------------
typedef unsigned short u16;
typedef unsigned int   u32;
typedef __attribute__((ext_vector_type(16))) __bf16 v16bf;
typedef __attribute__((ext_vector_type(8)))  float  v8f;

#define B_    2
#define S_    2048
#define HID_  4096
#define NH_   32
#define NKV_  8
#define HD_   128
#define MROWS (B_ * S_)      // 4096 flattened (b,s) rows
#define QCOLS (NH_ * HD_)    // 4096
#define KCOLS (NKV_ * HD_)   // 1024

union Frag16 {               // one WMMA 16-bit A/B operand (8 VGPRs)
  v16bf v;
  u32   u[8];
};

__device__ __forceinline__ u16 f2bf(float f) {
  u32 u = __builtin_bit_cast(u32, f);
  u += 0x7fffu + ((u >> 16) & 1u);    // round-to-nearest-even
  return (u16)(u >> 16);
}

__device__ __forceinline__ v8f wmma_bf16(const Frag16& a, const Frag16& b, v8f c) {
  // D(16x16,f32) = A(16x32,bf16) * B(32x16,bf16) + C
  return __builtin_amdgcn_wmma_f32_16x16x32_bf16(
      /*neg_a=*/false, a.v, /*neg_b=*/false, b.v,
      /*c_mod=*/(short)0, c, /*reuse_a=*/false, /*reuse_b=*/false);
}

// Fetch 4 consecutive K-elements from global as packed bf16 pair-of-dwords.
__device__ __forceinline__ uint2 fetch4(const float* __restrict__ g) {
  float4 f = *(const float4*)g;
  u32 lo = (u32)f2bf(f.x) | ((u32)f2bf(f.y) << 16);
  u32 hi = (u32)f2bf(f.z) | ((u32)f2bf(f.w) << 16);
  return make_uint2(lo, hi);
}
__device__ __forceinline__ uint2 fetch4(const u16* __restrict__ g) {
  return *(const uint2*)g;
}

// ---------------------------------------------------------------------------
// GEMM: C[M,N] (f32) = A[M,K] * W[N,K]^T, A is fp32 or bf16, W fp32.
// Block tile 128x128, 8 waves of 32 lanes; wave tile 32(M)x64(N); K step 64.
// Software-pipelined: tile k+1 is fetched from global into registers while
// the WMMAs consume tile k from LDS.
// LDS row stride 72 halfwords = 36 dwords -> 16B-aligned staging stores and
// bank-conflict-free fragment reads (36*r mod 64 distinct for r in [0,16)).
// ---------------------------------------------------------------------------
template <typename TA>
__device__ __forceinline__ void fetch_tiles(
    const TA* __restrict__ A, const float* __restrict__ W,
    int bm, int bn, int K, int k0, int tid, uint2* ra, uint2* rw)
{
#pragma unroll
  for (int p = 0; p < 8; ++p) {
    int r = p * 16 + (tid >> 4);
    int c = (tid & 15) * 4;
    ra[p] = fetch4(&A[(size_t)(bm + r) * K + k0 + c]);
    rw[p] = fetch4(&W[(size_t)(bn + r) * K + k0 + c]);
  }
}

template <typename TA>
__global__ __launch_bounds__(256) void gemm_wmma_bf16(
    const TA* __restrict__ A, const float* __restrict__ W,
    float* __restrict__ C, int M, int N, int K)
{
  __shared__ __align__(16) u16 As[128][72];
  __shared__ __align__(16) u16 Ws[128][72];

  const int tid  = threadIdx.x;
  const int lane = tid & 31;
  const int wid  = tid >> 5;
  const int wm   = (wid & 3) * 32;     // wave M origin in block tile
  const int wn   = (wid >> 2) * 64;    // wave N origin in block tile
  const int bm   = blockIdx.y * 128;
  const int bn   = blockIdx.x * 128;
  const int lh   = lane & 15;          // lane-in-16
  const int lx   = lane >> 4;          // lane half (K-split per ISA layout)

  v8f acc[2][4] = {};
  uint2 ra[8], rw[8];

  fetch_tiles(A, W, bm, bn, K, 0, tid, ra, rw);   // prologue fetch

  for (int k0 = 0; k0 < K; k0 += 64) {
    // --- commit the register-staged tile to LDS ---
#pragma unroll
    for (int p = 0; p < 8; ++p) {
      int r = p * 16 + (tid >> 4);
      int c = (tid & 15) * 4;
      *(uint2*)&As[r][c] = ra[p];
      *(uint2*)&Ws[r][c] = rw[p];
    }
    __syncthreads();

    // --- kick off the next tile's global fetch (overlaps the WMMAs) ---
    if (k0 + 64 < K)
      fetch_tiles(A, W, bm, bn, K, k0 + 64, tid, ra, rw);

#pragma unroll
    for (int kc = 0; kc < 2; ++kc) {   // two 32-wide K chunks
      Frag16 af[2], bf[4];
#pragma unroll
      for (int mi = 0; mi < 2; ++mi) { // A frag: lane<16 K{0..7,16..23} pairs
        const u32* rp = (const u32*)&As[wm + mi * 16 + lh][kc * 32];
        int off = lx * 4;
#pragma unroll
        for (int i = 0; i < 4; ++i) {
          af[mi].u[i]     = rp[off + i];
          af[mi].u[4 + i] = rp[off + 8 + i];
        }
      }
#pragma unroll
      for (int ni = 0; ni < 4; ++ni) { // B frag: col n=lh, K{0..7,16..23} pairs
        const u32* rp = (const u32*)&Ws[wn + ni * 16 + lh][0];
        int off = kc * 16 + lx * 4;
#pragma unroll
        for (int i = 0; i < 8; ++i) bf[ni].u[i] = rp[off + i + (i >= 4 ? 4 : 0)];
      }
#pragma unroll
      for (int mi = 0; mi < 2; ++mi)
#pragma unroll
        for (int ni = 0; ni < 4; ++ni)
          acc[mi][ni] = wmma_bf16(af[mi], bf[ni], acc[mi][ni]);
    }
    __syncthreads();
  }

  // epilogue: C/D layout -> lane holds column n=lh, rows i + 8*lx
#pragma unroll
  for (int mi = 0; mi < 2; ++mi)
#pragma unroll
    for (int ni = 0; ni < 4; ++ni) {
      int n  = bn + wn + ni * 16 + lh;
      int m0 = bm + wm + mi * 16 + 8 * lx;
#pragma unroll
      for (int i = 0; i < 8; ++i)
        C[(size_t)(m0 + i) * N + n] = acc[mi][ni][i];
    }
}

// ---------------------------------------------------------------------------
// RoPE (half-pair per thread) + fp32 -> bf16.  Layout: [row=b*S+s][h*128+d].
// q'[d]     = q[d]*cos - q[d+64]*sin
// q'[d+64]  = q[d+64]*cos + q[d]*sin,  freq index = d (d<64)
// ---------------------------------------------------------------------------
__global__ __launch_bounds__(256) void rope_to_bf16(
    const float* __restrict__ X, const int* __restrict__ pos,
    u16* __restrict__ O, int nheads)
{
  int idx     = blockIdx.x * 256 + threadIdx.x;
  int per_row = nheads * 64;
  int row     = idx / per_row;
  int rem     = idx - row * per_row;
  int h       = rem >> 6;
  int i       = rem & 63;
  int stride  = nheads * 128;

  float p    = (float)pos[row];
  // inv_freq = 10000^(-2i/128); log2(10000) = 13.287712379549449
  float invf = exp2f(-(float)(2 * i) * (13.28771237954945f / 128.0f));
  float ang  = p * invf;
  float c = cosf(ang), s = sinf(ang);

  size_t base = (size_t)row * stride + (size_t)h * 128 + i;
  float x0 = X[base], x1 = X[base + 64];
  O[base]      = f2bf(x0 * c - x1 * s);
  O[base + 64] = f2bf(x1 * c + x0 * s);
}

__global__ __launch_bounds__(256) void conv_bf16(
    const float* __restrict__ X, u16* __restrict__ O)
{
  int idx = blockIdx.x * 256 + threadIdx.x;
  O[idx] = f2bf(X[idx]);
}

// ---------------------------------------------------------------------------
// Flash attention (causal, GQA 4:1). One block = (b, head, 64-query tile),
// 4 waves x 16 query rows. KV streamed in 64-key tiles; V staged transposed.
// Scores/out accumulate in f32 WMMA accumulators; softmax in log2 domain.
// Next KV tile is prefetched (global_prefetch_b8) during the compute phase.
// ---------------------------------------------------------------------------
__global__ __launch_bounds__(128) void flash_attn(
    const u16* __restrict__ Q, const u16* __restrict__ Kg,
    const u16* __restrict__ Vg, u16* __restrict__ O)
{
  __shared__ __align__(16) u16 Qs[64][136];   // 136 hw = 68 dw: conflict-free
  __shared__ __align__(16) u16 Ks[64][136];
  __shared__ __align__(16) u16 Vt[128][72];   // V transposed: [d][key]
  __shared__ __align__(16) u16 Ps[4][16][72]; // per-wave P scratch

  const int tid = threadIdx.x, lane = tid & 31, w = tid >> 5;
  const int lh = lane & 15, lx = lane >> 4;
  const int qt = blockIdx.x, h = blockIdx.y, b = blockIdx.z;
  const int g  = h >> 2;  // NREP = 4

  const size_t qbase = ((size_t)b * S_ + (size_t)qt * 64) * QCOLS + (size_t)h * HD_;
  const size_t kvb0  = (size_t)b * S_ * KCOLS + (size_t)g * HD_;

  // --- stage Q tile (64x128) ---
#pragma unroll
  for (int p = 0; p < 8; ++p) {
    int r = p * 8 + (tid >> 4);
    int ch = tid & 15;
    *(uint4*)&Qs[r][ch * 8] = *(const uint4*)(Q + qbase + (size_t)r * QCOLS + ch * 8);
  }
  __syncthreads();

  // --- Q A-fragments (16 rows x 128 dims = 4 chunks of K=32), kept in regs ---
  Frag16 qf[4];
  {
    const u32* rp = (const u32*)&Qs[w * 16 + lh][0];
    int off = lx * 4;
#pragma unroll
    for (int kc = 0; kc < 4; ++kc)
#pragma unroll
      for (int i = 0; i < 4; ++i) {
        qf[kc].u[i]     = rp[kc * 16 + off + i];
        qf[kc].u[4 + i] = rp[kc * 16 + 8 + off + i];
      }
  }

  v8f oacc[8] = {};
  float rm[8], rs[8];
#pragma unroll
  for (int i = 0; i < 8; ++i) { rm[i] = -1e30f; rs[i] = 0.0f; }

  const float sl2 = 0.08838834764831845f * 1.4426950408889634f; // scale*log2(e)

  for (int kv = 0; kv <= qt; ++kv) {   // causal: skip fully-masked tiles
    __syncthreads();
    const size_t kb = kvb0 + (size_t)kv * 64 * KCOLS;
    // stage K tile (64x128)
#pragma unroll
    for (int p = 0; p < 8; ++p) {
      int r = p * 8 + (tid >> 4);
      int ch = tid & 15;
      *(uint4*)&Ks[r][ch * 8] = *(const uint4*)(Kg + kb + (size_t)r * KCOLS + ch * 8);
    }
    // stage V tile transposed: Vt[d][key]
#pragma unroll
    for (int p = 0; p < 32; ++p) {
      int li = p * 128 + tid;
      int key = li >> 6, du = li & 63;
      u32 uv = *(const u32*)(Vg + kb + (size_t)key * KCOLS + du * 2);
      Vt[du * 2][key]     = (u16)(uv & 0xffffu);
      Vt[du * 2 + 1][key] = (u16)(uv >> 16);
    }
    __syncthreads();

    // --- prefetch next KV tile into cache while we compute this one ---
    if (kv < qt) {
      const size_t kbn = kb + (size_t)64 * KCOLS;
      const u16* pk = Kg + kbn + (size_t)(tid & 63) * KCOLS + (tid >> 6) * 64;
      const u16* pv = Vg + kbn + (size_t)(tid & 63) * KCOLS + (tid >> 6) * 64;
      __builtin_prefetch(pk, 0, 1);
      __builtin_prefetch(pv, 0, 1);
    }

    // --- scores S(16x64) = Q(16x128) * K^T ---
    v8f sacc[4] = {};
#pragma unroll
    for (int kc = 0; kc < 4; ++kc) {
#pragma unroll
      for (int ni = 0; ni < 4; ++ni) {
        Frag16 kf;
        const u32* rp = (const u32*)&Ks[ni * 16 + lh][0];
        int off = kc * 16 + lx * 4;
#pragma unroll
        for (int i = 0; i < 8; ++i) kf.u[i] = rp[off + i + (i >= 4 ? 4 : 0)];
        sacc[ni] = wmma_bf16(qf[kc], kf, sacc[ni]);
      }
    }

    // --- scale to log2 domain + causal mask on the diagonal tile ---
    const bool diag = (kv == qt);
#pragma unroll
    for (int ni = 0; ni < 4; ++ni)
#pragma unroll
      for (int i = 0; i < 8; ++i) {
        float s = sacc[ni][i] * sl2;
        if (diag) {
          int qi = w * 16 + 8 * lx + i;
          int kj = ni * 16 + lh;
          if (kj > qi) s = -1e30f;
        }
        sacc[ni][i] = s;
      }

    // --- online softmax (rows live in frag index i + lane half lx) ---
#pragma unroll
    for (int i = 0; i < 8; ++i) {
      float m = sacc[0][i];
#pragma unroll
      for (int ni = 1; ni < 4; ++ni) m = fmaxf(m, sacc[ni][i]);
      m = fmaxf(m, __shfl_xor(m, 1));
      m = fmaxf(m, __shfl_xor(m, 2));
      m = fmaxf(m, __shfl_xor(m, 4));
      m = fmaxf(m, __shfl_xor(m, 8));   // xor<=8 stays within the 16-lane half
      float mn = fmaxf(rm[i], m);
      float sum = 0.0f;
#pragma unroll
      for (int ni = 0; ni < 4; ++ni) {
        float pr = exp2f(sacc[ni][i] - mn);
        sacc[ni][i] = pr;               // sacc now holds probabilities
        sum += pr;
      }
      sum += __shfl_xor(sum, 1);
      sum += __shfl_xor(sum, 2);
      sum += __shfl_xor(sum, 4);
      sum += __shfl_xor(sum, 8);
      float alpha = exp2f(rm[i] - mn);
      rs[i] = rs[i] * alpha + sum;
      rm[i] = mn;
#pragma unroll
      for (int ni = 0; ni < 8; ++ni) oacc[ni][i] *= alpha;
    }

    // --- P (bf16) -> per-wave LDS scratch (same-wave DS ordering, no barrier)
#pragma unroll
    for (int ni = 0; ni < 4; ++ni)
#pragma unroll
      for (int i = 0; i < 8; ++i)
        Ps[w][8 * lx + i][ni * 16 + lh] = f2bf(sacc[ni][i]);

    // --- O(16x128) += P(16x64) * V(64x128) ---
#pragma unroll
    for (int kc = 0; kc < 2; ++kc) {
      Frag16 pf;
      {
        const u32* rp = (const u32*)&Ps[w][lh][0];
        int off = lx * 4;
#pragma unroll
        for (int i = 0; i < 4; ++i) {
          pf.u[i]     = rp[kc * 16 + off + i];
          pf.u[4 + i] = rp[kc * 16 + 8 + off + i];
        }
      }
#pragma unroll
      for (int ni = 0; ni < 8; ++ni) {
        Frag16 vf;
        const u32* rp = (const u32*)&Vt[ni * 16 + lh][0];
        int off = kc * 16 + lx * 4;
#pragma unroll
        for (int i = 0; i < 8; ++i) vf.u[i] = rp[off + i + (i >= 4 ? 4 : 0)];
        oacc[ni] = wmma_bf16(pf, vf, oacc[ni]);
      }
    }
  }

  // --- normalize and emit bf16 in (b, s, NH*HD) layout for output proj ---
  const size_t obase = qbase + (size_t)(w * 16) * QCOLS;
#pragma unroll
  for (int ni = 0; ni < 8; ++ni)
#pragma unroll
    for (int i = 0; i < 8; ++i) {
      int m = 8 * lx + i;
      O[obase + (size_t)m * QCOLS + ni * 16 + lh] = f2bf(oacc[ni][i] / rs[i]);
    }
}

// ---------------------------------------------------------------------------
// Launch: QKV GEMMs -> RoPE/convert -> flash attention -> output GEMM.
// ---------------------------------------------------------------------------
extern "C" void kernel_launch(void* const* d_in, const int* in_sizes, int n_in,
                              void* d_out, int out_size, void* d_ws, size_t ws_size,
                              hipStream_t stream) {
  (void)in_sizes; (void)n_in; (void)out_size; (void)ws_size;
  const float* hs = (const float*)d_in[0];
  const float* wq = (const float*)d_in[1];
  const float* wk = (const float*)d_in[2];
  const float* wv = (const float*)d_in[3];
  const float* wo = (const float*)d_in[4];
  // d_in[5] = attention_mask (pure causal; applied analytically in-kernel)
  const int* pos  = (const int*)d_in[6];
  float* out = (float*)d_out;

  // workspace carve-up (~184 MB)
  char* ws = (char*)d_ws;
  float* qf = (float*)(ws);                               // 4096x4096 f32
  float* kf = (float*)(ws + (size_t)67108864);            // 4096x1024 f32
  float* vf = (float*)(ws + (size_t)83886080);            // 4096x1024 f32
  u16*   qb = (u16*)  (ws + (size_t)100663296);           // 4096x4096 bf16
  u16*   kb = (u16*)  (ws + (size_t)134217728);           // 4096x1024 bf16
  u16*   vb = (u16*)  (ws + (size_t)142606336);           // 4096x1024 bf16
  u16*   ob = (u16*)  (ws + (size_t)150994944);           // 4096x4096 bf16

  // QKV projections (fp32 A/W staged as bf16, f32 WMMA accumulate)
  gemm_wmma_bf16<float><<<dim3(QCOLS / 128, MROWS / 128), 256, 0, stream>>>(
      hs, wq, qf, MROWS, QCOLS, HID_);
  gemm_wmma_bf16<float><<<dim3(KCOLS / 128, MROWS / 128), 256, 0, stream>>>(
      hs, wk, kf, MROWS, KCOLS, HID_);
  gemm_wmma_bf16<float><<<dim3(KCOLS / 128, MROWS / 128), 256, 0, stream>>>(
      hs, wv, vf, MROWS, KCOLS, HID_);

  // RoPE for q/k, plain convert for v (all -> bf16)
  rope_to_bf16<<<(MROWS * NH_ * 64) / 256, 256, 0, stream>>>(qf, pos, qb, NH_);
  rope_to_bf16<<<(MROWS * NKV_ * 64) / 256, 256, 0, stream>>>(kf, pos, kb, NKV_);
  conv_bf16<<<(MROWS * KCOLS) / 256, 256, 0, stream>>>(vf, vb);

  // causal flash attention
  flash_attn<<<dim3(S_ / 64, NH_, B_), 128, 0, stream>>>(qb, kb, vb, ob);

  // output projection -> fp32 result
  gemm_wmma_bf16<u16><<<dim3(HID_ / 128, MROWS / 128), 256, 0, stream>>>(
      ob, wo, out, MROWS, HID_, QCOLS);
}
